// ConsensusTransportMemoryPool_83958020702564
// MI455X (gfx1250) — matrix operand order repaired and checked by reading, not compiled, not run on hardware
//
#include <hip/hip_runtime.h>
#include <hip/hip_bf16.h>

// ---- CDNA5 WMMA types ------------------------------------------------------
typedef __attribute__((ext_vector_type(16))) __bf16 v16bf;
typedef __attribute__((ext_vector_type(8)))  __bf16 v8bf;
typedef __attribute__((ext_vector_type(8)))  float  v8f;

union V16U { v16bf v; v8bf h[2]; };

#define DIM   1024
#define TP    8192      // tokens per way (16 shots * 512)
#define LDA   1032      // padded bf16 LDS row stride (16B-aligned, bank-spread)
#define LDP   72        // padded P-tile bf16 stride
#define LDL   68        // padded logits f32 stride

__device__ __forceinline__ unsigned short f2bf(float f) {
  unsigned u = __builtin_bit_cast(unsigned, f);
  return (unsigned short)((u + 0x7FFFu + ((u >> 16) & 1u)) >> 16);  // RNE
}

__device__ __forceinline__ unsigned pack_bf2(float lo, float hi) {
  return (unsigned)f2bf(lo) | ((unsigned)f2bf(hi) << 16);
}

__device__ __forceinline__ v8f wmma_bf16(const v16bf& a, const v16bf& b, const v8f& c) {
  return __builtin_amdgcn_wmma_f32_16x16x32_bf16(false, a, false, b, (short)0, c, false, false);
}

// ---- kernel 1: f32 -> bf16 weight conversion -------------------------------
__global__ void ctmp_cvt_bf16(const float* __restrict__ in,
                              unsigned short* __restrict__ out, int n) {
  int i = blockIdx.x * 256 + threadIdx.x;
  if (i < n) out[i] = f2bf(in[i]);
}

// ---- kernel 2: fused LayerNorm + GEMM  out = LN(X) @ W^T  (bf16 WMMA) ------
// X: (W, rows, D) f32.  Wb: (D, D) bf16 row-major (row n = output feature n).
// TRANSPOSED==0: out[way][row][d]; TRANSPOSED==1: out[way][d][row] (row-stride=rows)
template <int TRANSPOSED, int USE_LN>
__global__ __launch_bounds__(256) void ctmp_proj(
    const float* __restrict__ X, int rows,
    const unsigned short* __restrict__ Wb,
    unsigned short* __restrict__ Out,
    const float* __restrict__ gamma, const float* __restrict__ beta) {
  __shared__ __align__(32) unsigned short lds_a[64 * LDA];  // 132 KB bf16 tile
  __shared__ float red[64][4][2];
  __shared__ float s_mu[64], s_rs[64];

  const int way = blockIdx.y;
  const int t0  = blockIdx.x * 64;
  const int tid = threadIdx.x;
  const float* Xw = X + ((size_t)way * rows + t0) * DIM;

  if (USE_LN) {                                   // per-row mean / rsqrt(var)
    int r = tid >> 2, seg = tid & 3;
    const float* xr = Xw + (size_t)r * DIM + seg * 256;
    float s = 0.f, ss = 0.f;
    for (int i = 0; i < 256; ++i) { float v = xr[i]; s += v; ss += v * v; }
    red[r][seg][0] = s; red[r][seg][1] = ss;
    __syncthreads();
    if (tid < 64) {
      float s0 = 0.f, q0 = 0.f;
      for (int j = 0; j < 4; ++j) { s0 += red[tid][j][0]; q0 += red[tid][j][1]; }
      float mu = s0 * (1.f / DIM);
      float var = q0 * (1.f / DIM) - mu * mu;
      s_mu[tid] = mu; s_rs[tid] = rsqrtf(var + 1e-5f);
    }
    __syncthreads();
  }
  for (int idx = tid; idx < 32 * DIM; idx += 256) {  // normalize + cvt to LDS
    int r = idx >> 9, d = (idx & 511) * 2;           // 2 elements per thread-iter
    float v0 = Xw[(size_t)r * DIM + d];
    float v1 = Xw[(size_t)r * DIM + d + 1];
    if (USE_LN) {
      float mu = s_mu[r], rs = s_rs[r];
      v0 = (v0 - mu) * rs * gamma[d] + beta[d];
      v1 = (v1 - mu) * rs * gamma[d + 1] + beta[d + 1];
    }
    *(unsigned*)&lds_a[r * LDA + d] = pack_bf2(v0, v1);
  }
  __syncthreads();

  const int wave = tid >> 5, lane = tid & 31;
  const int lane15 = lane & 15, hi = lane >> 4;
  const v8f vz = {0.f, 0.f, 0.f, 0.f, 0.f, 0.f, 0.f, 0.f};

  for (int ns = 0; ns < 8; ++ns) {                // wave owns 128 out-columns
    int n0 = wave * 128 + ns * 16;
    v8f acc[4]; acc[0] = vz; acc[1] = vz; acc[2] = vz; acc[3] = vz;
    const unsigned short* wrow = Wb + (size_t)(n0 + lane15) * DIM;
    for (int kk = 0; kk < 32; ++kk) {             // K = 1024, 32 per wmma
      int k0 = kk * 32;
      v16bf b = *(const v16bf*)(wrow + k0 + hi * 16);
#pragma unroll
      for (int ms = 0; ms < 4; ++ms) {
        const unsigned short* ar = lds_a + (ms * 16 + lane15) * LDA + k0 + hi * 16;
        V16U a; a.h[0] = *(const v8bf*)ar; a.h[1] = *(const v8bf*)(ar + 8);
        acc[ms] = wmma_bf16(a.v, b, acc[ms]);
      }
    }
    const int n = n0 + lane15;
#pragma unroll
    for (int ms = 0; ms < 4; ++ms) {              // store C (bf16)
      int mb = ms * 16 + hi * 8;
      if (TRANSPOSED) {
        // 8 consecutive t-elements per lane -> one 16B store
        unsigned pk[4];
#pragma unroll
        for (int p = 0; p < 4; ++p) pk[p] = pack_bf2(acc[ms][2 * p], acc[ms][2 * p + 1]);
        uint4 w4 = {pk[0], pk[1], pk[2], pk[3]};
        *(uint4*)&Out[((size_t)way * DIM + n) * rows + t0 + mb] = w4;
      } else {
#pragma unroll
        for (int j = 0; j < 8; ++j)
          Out[((size_t)way * rows + t0 + mb + j) * DIM + n] = f2bf(acc[ms][j]);
      }
    }
  }
}

// ---- kernel 3: flash attention + residual + output LayerNorm ---------------
// grid (8, 16): blockIdx.x = 32-slot block, blockIdx.y = way. 256 thr = 8 waves.
__global__ __launch_bounds__(256) void ctmp_attn(
    const float* __restrict__ cs, const float* __restrict__ sw,
    const unsigned short* __restrict__ qb, const unsigned short* __restrict__ kb,
    const unsigned short* __restrict__ vtb,
    const float* __restrict__ g_out, const float* __restrict__ b_out,
    float* __restrict__ out) {
  __shared__ __align__(32) unsigned short lds_q[32 * LDA];   // 66 KB
  __shared__ __align__(32) unsigned short lds_p[32 * LDP];   // bf16 P tile
  __shared__ float lds_logit[32 * LDL];
  __shared__ float s_scale[32], s_invl[32], s_bias[16];
  __shared__ float s_stage[32 * 132];
  __shared__ float s_red[256][2];
  __shared__ float s_mu[32], s_rs[32];

  const int way = blockIdx.y, s0 = blockIdx.x * 32;
  const int tid = threadIdx.x;

  const unsigned short* qsrc = qb + ((size_t)way * 256 + s0) * DIM;
  for (int idx = tid; idx < 32 * (DIM / 4); idx += 256) {    // 8B vector copy
    int r = idx >> 8, d = (idx & 255) * 4;
    *(uint2*)&lds_q[r * LDA + d] = *(const uint2*)&qsrc[(size_t)r * DIM + d];
  }
  if (tid < 16) s_bias[tid] = logf(fmaxf(sw[way * 16 + tid], 1e-6f));
  __syncthreads();

  const int wave = tid >> 5, lane = tid & 31;
  const int lane15 = lane & 15, hi = lane >> 4;
  const int msl = wave & 1, tn = wave >> 1;       // logits tile owned by wave
  const float scale = 0.03125f;                   // 1/sqrt(1024)
  const v8f vz = {0.f, 0.f, 0.f, 0.f, 0.f, 0.f, 0.f, 0.f};

  v8f acc[2][8];                                  // O accumulator: 128 d-cols
#pragma unroll
  for (int ms = 0; ms < 2; ++ms)
#pragma unroll
    for (int ns = 0; ns < 8; ++ns) acc[ms][ns] = vz;
  float m_run = -3.4e38f, l_run = 0.f;            // rows owned by tid<32

  const unsigned short* kw = kb  + (size_t)way * TP * DIM;
  const unsigned short* vw = vtb + (size_t)way * DIM * TP;

  for (int t0 = 0; t0 < TP; t0 += 64) {           // --- flash loop ---
    // logits tile (16x16), K=1024
    v8f lacc = vz;
    int tcol = t0 + tn * 16 + lane15;
    const unsigned short* krow = kw + (size_t)tcol * DIM;
    if (t0 + 64 < TP) {                           // prefetch next K rows
      __builtin_prefetch(krow + (size_t)64 * DIM, 0, 1);
      __builtin_prefetch(krow + (size_t)64 * DIM + 512, 0, 1);
    }
    for (int kk = 0; kk < 32; ++kk) {
      int k0 = kk * 32;
      v16bf b = *(const v16bf*)(krow + k0 + hi * 16);
      const unsigned short* ar = lds_q + (msl * 16 + lane15) * LDA + k0 + hi * 16;
      V16U a; a.h[0] = *(const v8bf*)ar; a.h[1] = *(const v8bf*)(ar + 8);
      lacc = wmma_bf16(a.v, b, lacc);
    }
    float bias = s_bias[tcol >> 9];               // 512 tokens per shot
    int col = tn * 16 + lane15;
#pragma unroll
    for (int j = 0; j < 8; ++j)
      lds_logit[(msl * 16 + hi * 8 + j) * LDL + col] = lacc[j] * scale + bias;
    __syncthreads();

    if (tid < 32) {                               // online softmax, row = tid
      int r = tid;
      float mx = m_run;
      for (int j = 0; j < 64; ++j) mx = fmaxf(mx, lds_logit[r * LDL + j]);
      float f = __expf(m_run - mx);
      float s = 0.f;
      for (int j = 0; j < 32; ++j) {
        float p0 = __expf(lds_logit[r * LDL + 2 * j] - mx);
        float p1 = __expf(lds_logit[r * LDL + 2 * j + 1] - mx);
        s += p0 + p1;
        *(unsigned*)&lds_p[r * LDP + 2 * j] = pack_bf2(p0, p1);
      }
      l_run = l_run * f + s;
      m_run = mx;
      s_scale[r] = f;
    }
    __syncthreads();

#pragma unroll
    for (int ms = 0; ms < 2; ++ms)                // rescale running O
#pragma unroll
      for (int ns = 0; ns < 8; ++ns)
#pragma unroll
        for (int j = 0; j < 8; ++j)
          acc[ms][ns][j] *= s_scale[ms * 16 + hi * 8 + j];

#pragma unroll
    for (int ns = 0; ns < 8; ++ns) {              // O += P @ V  (K = 64 tokens)
      int dcol = wave * 128 + ns * 16 + lane15;
      const unsigned short* vrow = vw + (size_t)dcol * TP + t0;
      if (t0 + 64 < TP) __builtin_prefetch(vrow + 64, 0, 1);
#pragma unroll
      for (int ts = 0; ts < 2; ++ts) {
        v16bf b = *(const v16bf*)(vrow + ts * 32 + hi * 16);
#pragma unroll
        for (int ms = 0; ms < 2; ++ms) {
          const unsigned short* ar = lds_p + (ms * 16 + lane15) * LDP + ts * 32 + hi * 16;
          V16U a; a.h[0] = *(const v8bf*)ar; a.h[1] = *(const v8bf*)(ar + 8);
          acc[ms][ns] = wmma_bf16(a.v, b, acc[ms][ns]);
        }
      }
    }
  }

  if (tid < 32) s_invl[tid] = 1.0f / l_run;
  __syncthreads();

  // epilogue: residual + output LayerNorm (cross-wave reduction via LDS)
  float psum = 0.f, psq = 0.f;
  const int rr = tid >> 3, ccc = tid & 7;
  for (int ns = 0; ns < 8; ++ns) {
#pragma unroll
    for (int ms = 0; ms < 2; ++ms)
#pragma unroll
      for (int j = 0; j < 8; ++j) {
        int m = ms * 16 + hi * 8 + j;
        int d = wave * 128 + ns * 16 + lane15;
        float val = cs[((size_t)way * 256 + s0 + m) * DIM + d] + acc[ms][ns][j] * s_invl[m];
        s_stage[m * 132 + wave * 16 + lane15] = val;
      }
    __syncthreads();
#pragma unroll
    for (int i = 0; i < 16; ++i) {
      float v = s_stage[rr * 132 + ccc * 16 + i];
      psum += v; psq += v * v;
    }
    __syncthreads();
  }
  s_red[tid][0] = psum; s_red[tid][1] = psq;
  __syncthreads();
  if (tid < 32) {
    float s = 0.f, q = 0.f;
    for (int c = 0; c < 8; ++c) { s += s_red[tid * 8 + c][0]; q += s_red[tid * 8 + c][1]; }
    float mu = s * (1.f / DIM), var = q * (1.f / DIM) - mu * mu;
    s_mu[tid] = mu; s_rs[tid] = rsqrtf(var + 1e-5f);
  }
  __syncthreads();
  for (int ns = 0; ns < 8; ++ns)
#pragma unroll
    for (int ms = 0; ms < 2; ++ms)
#pragma unroll
      for (int j = 0; j < 8; ++j) {
        int m = ms * 16 + hi * 8 + j;
        int d = wave * 128 + ns * 16 + lane15;
        size_t gi = ((size_t)way * 256 + s0 + m) * DIM + d;
        float val = cs[gi] + acc[ms][ns][j] * s_invl[m];
        out[gi] = (val - s_mu[m]) * s_rs[m] * g_out[d] + b_out[d];
      }
}

// ---- host launch -----------------------------------------------------------
extern "C" void kernel_launch(void* const* d_in, const int* in_sizes, int n_in,
                              void* d_out, int out_size, void* d_ws, size_t ws_size,
                              hipStream_t stream) {
  const float* cs   = (const float*)d_in[0];   // (16,256,1024)
  const float* st   = (const float*)d_in[1];   // (16,16,512,1024)
  const float* sw   = (const float*)d_in[2];   // (16,16)
  const float* Wq   = (const float*)d_in[3];
  const float* Wk   = (const float*)d_in[4];
  const float* Wv   = (const float*)d_in[5];
  const float* g_slot  = (const float*)d_in[6];
  const float* b_slot  = (const float*)d_in[7];
  const float* g_trans = (const float*)d_in[8];
  const float* b_trans = (const float*)d_in[9];
  const float* g_out   = (const float*)d_in[10];
  const float* b_out   = (const float*)d_in[11];

  // workspace layout (bf16): 3 weights (2MB ea) | q (8MB) | k (256MB) | vT (256MB)
  unsigned short* Wq_b = (unsigned short*)d_ws;
  unsigned short* Wk_b = Wq_b + (size_t)DIM * DIM;
  unsigned short* Wv_b = Wk_b + (size_t)DIM * DIM;
  unsigned short* q_b  = Wv_b + (size_t)DIM * DIM;
  unsigned short* k_b  = q_b  + (size_t)16 * 256 * DIM;
  unsigned short* vT_b = k_b  + (size_t)16 * TP * DIM;

  ctmp_cvt_bf16<<<4096, 256, 0, stream>>>(Wq, Wq_b, DIM * DIM);
  ctmp_cvt_bf16<<<4096, 256, 0, stream>>>(Wk, Wk_b, DIM * DIM);
  ctmp_cvt_bf16<<<4096, 256, 0, stream>>>(Wv, Wv_b, DIM * DIM);

  // q = LN(class_slots) @ Wq^T    (row-major bf16)
  ctmp_proj<0, 1><<<dim3(4, 16), 256, 0, stream>>>(cs, 256, Wq_b, q_b, g_slot, b_slot);
  // k = LN(flat) @ Wk^T           (row-major bf16)
  ctmp_proj<0, 1><<<dim3(128, 16), 256, 0, stream>>>(st, TP, Wk_b, k_b, g_trans, b_trans);
  // v = flat @ Wv^T               (stored transposed: vT[way][d][t])
  ctmp_proj<1, 0><<<dim3(128, 16), 256, 0, stream>>>(st, TP, Wv_b, vT_b, nullptr, nullptr);

  // flash attention + residual + output LN
  ctmp_attn<<<dim3(8, 16), 256, 0, stream>>>(cs, sw, q_b, k_b, vT_b, g_out, b_out,
                                             (float*)d_out);
}
